// MultiViewEvidenceRouter_51634096832664
// MI455X (gfx1250) — matrix operand order repaired and checked by reading, not compiled, not run on hardware
//
#include <hip/hip_runtime.h>
#include <hip/hip_bf16.h>
#include <math.h>

#define BB 8
#define QQ 64
#define KK 8192
#define HH 1024
#define NN 512
#define CBLK 32       // context block size for prefix (small => tiny residual loops)
#define CSH  5        // log2(CBLK)
#define NBLK 256      // KK / CBLK
#define LN_EPS 1e-5f

typedef __attribute__((ext_vector_type(16))) __bf16 v16bf;
typedef __attribute__((ext_vector_type(8)))  float  v8f;

union FragBF16 { v16bf v; unsigned int u[8]; uint4 q[2]; };

// ---------------- block reduction helper (power-of-two blockDim) --------------
__device__ __forceinline__ float block_reduce_sum(float v, float* sbuf) {
    const int tid = threadIdx.x;
    sbuf[tid] = v;
    __syncthreads();
    for (int s = blockDim.x >> 1; s > 0; s >>= 1) {
        if (tid < s) sbuf[tid] += sbuf[tid + s];
        __syncthreads();
    }
    float r = sbuf[0];
    __syncthreads();
    return r;
}

// ---------------- 1) q_vec = LN(mean over Q)  -> [B,H] f32 --------------------
__global__ void mver_qvec_kernel(const float* __restrict__ q_emb,
                                 const float* __restrict__ ln_w,
                                 const float* __restrict__ ln_b,
                                 float* __restrict__ q_vec) {
    __shared__ float sbuf[256];
    const int b = blockIdx.x;
    const int tid = threadIdx.x;
    float x[4];
    float lsum = 0.f, lsq = 0.f;
    #pragma unroll
    for (int i = 0; i < 4; ++i) {
        const int h = tid + 256 * i;
        float s = 0.f;
        for (int q = 0; q < QQ; ++q)
            s += q_emb[((size_t)b * QQ + q) * HH + h];
        x[i] = s * (1.0f / (float)QQ);
        lsum += x[i];
        lsq  += x[i] * x[i];
    }
    const float mu  = block_reduce_sum(lsum, sbuf) * (1.0f / (float)HH);
    const float ex2 = block_reduce_sum(lsq,  sbuf) * (1.0f / (float)HH);
    const float rstd = rsqrtf(ex2 - mu * mu + LN_EPS);
    #pragma unroll
    for (int i = 0; i < 4; ++i) {
        const int h = tid + 256 * i;
        q_vec[(size_t)b * HH + h] = (x[i] - mu) * rstd * ln_w[h] + ln_b[h];
    }
}

// ---------------- 2) per-32-row block sums of context  bs[B,256,H] ------------
__global__ void mver_blocksum_kernel(const float* __restrict__ ctx,
                                     float* __restrict__ bs) {
    int t = blockIdx.x;
    const int hc = t & 3; t >>= 2;
    const int j  = t & (NBLK - 1); t >>= 8;
    const int b  = t;
    const int h  = hc * 256 + threadIdx.x;
    float acc = 0.f;
    const size_t base = ((size_t)b * KK + (size_t)j * CBLK) * HH + h;
    for (int k = 0; k < CBLK; ++k)
        acc += ctx[base + (size_t)k * HH];
    bs[((size_t)b * NBLK + j) * HH + h] = acc;
}

// ---------------- 3) exclusive prefix over blocks  bp[B,257,H] ----------------
__global__ void mver_blockprefix_kernel(const float* __restrict__ bs,
                                        float* __restrict__ bp) {
    const int b = blockIdx.x >> 2;
    const int h = (blockIdx.x & 3) * 256 + threadIdx.x;
    float run = 0.f;
    bp[((size_t)b * (NBLK + 1)) * HH + h] = 0.f;
    for (int j = 0; j < NBLK; ++j) {
        run += bs[((size_t)b * NBLK + j) * HH + h];
        bp[((size_t)b * (NBLK + 1) + j + 1) * HH + h] = run;
    }
}

// ---------------- 4) W1 bottom half -> bf16, transposed to [j][h] -------------
__global__ void mver_w1t_kernel(const float* __restrict__ w1,
                                __hip_bfloat16* __restrict__ w1T) {
    const int idx = blockIdx.x * 256 + threadIdx.x; // j*HH + h
    const int j  = idx >> 10;
    const int hh = idx & (HH - 1);
    w1T[idx] = __float2bfloat16(w1[(size_t)(HH + hh) * HH + j]);
}

// ---------------- 5) qW1[b][j] = b1[j] + q_vec[b]·W1_top[:,j] -----------------
__global__ void mver_qw1_kernel(const float* __restrict__ q_vec,
                                const float* __restrict__ w1,
                                const float* __restrict__ b1,
                                float* __restrict__ qW1) {
    const int b = blockIdx.x >> 2;
    const int j = (blockIdx.x & 3) * 256 + threadIdx.x;
    float acc = b1[j];
    for (int hh = 0; hh < HH; ++hh)
        acc += q_vec[(size_t)b * HH + hh] * w1[(size_t)hh * HH + j];
    qW1[(size_t)b * HH + j] = acc;
}

// ---------------- 6) span seg-mean + LN -> chunk bf16 [B*N, H] ----------------
__global__ void mver_span_kernel(const float* __restrict__ ctx,
                                 const int* __restrict__ spans,
                                 const float* __restrict__ bp,
                                 const float* __restrict__ ln_w,
                                 const float* __restrict__ ln_b,
                                 __hip_bfloat16* __restrict__ chunk) {
    __shared__ float sbuf[256];
    const int bn = blockIdx.x;
    const int b  = bn >> 9;           // / NN
    const int tid = threadIdx.x;
    int s = spans[(size_t)bn * 2 + 0];
    int e = spans[(size_t)bn * 2 + 1];
    s = min(max(s, 0), KK);
    e = min(max(e, s), KK);
    const int len = e - s;
    const int sb = s >> CSH, eb = e >> CSH;
    const int h = tid * 4;

    const float4* bpv = (const float4*)bp;
    float4 pe = bpv[((size_t)(b * (NBLK + 1) + eb) * HH + h) >> 2];
    float4 ps = bpv[((size_t)(b * (NBLK + 1) + sb) * HH + h) >> 2];
    float ax = pe.x - ps.x, ay = pe.y - ps.y, az = pe.z - ps.z, aw = pe.w - ps.w;

    for (int k = sb * CBLK; k < s; ++k) {
        const float4 r = ((const float4*)(ctx + ((size_t)b * KK + k) * HH))[tid];
        ax -= r.x; ay -= r.y; az -= r.z; aw -= r.w;
    }
    for (int k = eb * CBLK; k < e; ++k) {
        const float4 r = ((const float4*)(ctx + ((size_t)b * KK + k) * HH))[tid];
        ax += r.x; ay += r.y; az += r.z; aw += r.w;
    }
    const float inv = 1.0f / (float)max(len, 1);
    ax *= inv; ay *= inv; az *= inv; aw *= inv;

    const float lsum = ax + ay + az + aw;
    const float lsq  = ax * ax + ay * ay + az * az + aw * aw;
    const float mu  = block_reduce_sum(lsum, sbuf) * (1.0f / (float)HH);
    const float ex2 = block_reduce_sum(lsq,  sbuf) * (1.0f / (float)HH);
    const float rstd = rsqrtf(ex2 - mu * mu + LN_EPS);
    const bool valid = (len > 0);

    float vals[4] = {ax, ay, az, aw};
    #pragma unroll
    for (int i = 0; i < 4; ++i) {
        const float o = valid ? (vals[i] - mu) * rstd * ln_w[h + i] + ln_b[h + i] : 0.0f;
        chunk[(size_t)bn * HH + h + i] = __float2bfloat16(o);
    }
}

// ---------------- 7) WMMA GEMM: hdn = chunk(bf16) x W1_bot^T + qW1 ------------
// M=4096 (b*N+n), N=1024 (j), K=1024 (h). Each wave -> 32x64 tile (2x4 WMMA).
// Each fragment = two contiguous b128 loads from a per-stream base pointer.
__global__ __launch_bounds__(256)
void mver_gemm_wmma_kernel(const __hip_bfloat16* __restrict__ A,   // [4096,1024] row-major bf16
                           const __hip_bfloat16* __restrict__ Bt,  // [1024(j),1024(h)] bf16
                           const float* __restrict__ qW1,          // [B,1024] f32 (includes b1)
                           float* __restrict__ hdn) {              // [4096,1024] f32
    const int lane = threadIdx.x & 31;
    const int wave = threadIdx.x >> 5;
    const int w = blockIdx.x * 8 + wave;            // 2048 waves
    const int tilesN = HH / 64;                     // 16
    const int tM = (w / tilesN) * 32;
    const int tN = (w % tilesN) * 64;
    const int b  = tM >> 9;                          // tile never crosses batch (512 % 32 == 0)
    const int hi  = (lane >> 4) & 1;
    const int l15 = lane & 15;

    const unsigned int* Au = (const unsigned int*)A;   // bf16 pairs
    const unsigned int* Bu = (const unsigned int*)Bt;

    // Per-stream base pointers (uint granularity). A: +4 uints when hi half,
    // B: +4 uints when hi half. Each advances 16 uints (=32 bf16) per k-step.
    const unsigned int* pA[2];
    const unsigned int* pB[4];
    #pragma unroll
    for (int i = 0; i < 2; ++i)
        pA[i] = Au + (((size_t)(tM + 16 * i + l15) * HH) >> 1) + (hi ? 4 : 0);
    #pragma unroll
    for (int j = 0; j < 4; ++j)
        pB[j] = Bu + (((size_t)(tN + 16 * j + l15) * HH) >> 1) + (hi ? 4 : 0);

    v8f c[2][4];
    #pragma unroll
    for (int j = 0; j < 4; ++j) {
        const float ci = qW1[(size_t)b * HH + tN + 16 * j + l15];
        #pragma unroll
        for (int i = 0; i < 2; ++i)
            #pragma unroll
            for (int r = 0; r < 8; ++r) c[i][j][r] = ci;
    }

    for (int k0 = 0; k0 < HH; k0 += 32) {
        FragBF16 a[2], bb[4];
        // A frag (ISA 16-bit A 16x32 table): uints {0,1,2,3} and {8,9,10,11} from base
        #pragma unroll
        for (int i = 0; i < 2; ++i) {
            a[i].q[0] = *(const uint4*)(pA[i]);
            a[i].q[1] = *(const uint4*)(pA[i] + 8);
            pA[i] += 16;
        }
        // B frag (same k order as A): uints {0,1,2,3} and {8,9,10,11} from base
        #pragma unroll
        for (int j = 0; j < 4; ++j) {
            bb[j].q[0] = *(const uint4*)(pB[j]);
            bb[j].q[1] = *(const uint4*)(pB[j] + 8);
            pB[j] += 16;
        }
        #pragma unroll
        for (int i = 0; i < 2; ++i)
            #pragma unroll
            for (int j = 0; j < 4; ++j)
                c[i][j] = __builtin_amdgcn_wmma_f32_16x16x32_bf16(
                    false, a[i].v, false, bb[j].v, (short)0, c[i][j], false, false);
    }

    // C/D layout: VGPR r -> M = r + (lane>=16 ? 8 : 0); N = lane&15 (per 16-wide tile)
    #pragma unroll
    for (int i = 0; i < 2; ++i) {
        const int mbase = tM + 16 * i + (hi ? 8 : 0);
        #pragma unroll
        for (int r = 0; r < 8; ++r)
            #pragma unroll
            for (int j = 0; j < 4; ++j)
                hdn[(size_t)(mbase + r) * HH + tN + 16 * j + l15] = c[i][j][r];
    }
}

// ---------------- 8) GELU(hdn)·w2 + b2 -> sigmoid -> fused --------------------
__global__ void mver_gate_kernel(const float* __restrict__ hdn,
                                 const float* __restrict__ w2,
                                 const float* __restrict__ b2,
                                 const float* __restrict__ local_s,
                                 const float* __restrict__ global_s,
                                 float* __restrict__ fused) {
    __shared__ float sbuf[256];
    const int bn = blockIdx.x;
    const int tid = threadIdx.x;
    float partial = 0.f;
    #pragma unroll
    for (int i = 0; i < 4; ++i) {
        const int h = tid + 256 * i;
        const float x = hdn[(size_t)bn * HH + h];
        const float g = 0.5f * x * (1.0f + erff(x * 0.70710678118654752f)); // exact GELU
        partial += g * w2[h];
    }
    const float dot = block_reduce_sum(partial, sbuf);
    if (tid == 0) {
        const float gate = 1.0f / (1.0f + expf(-(dot + b2[0])));
        fused[bn] = gate * local_s[bn] + (1.0f - gate) * global_s[bn];
    }
}

// ---------------- 9) softmax over N per batch + mean --------------------------
__global__ void mver_softmax_kernel(const float* __restrict__ fused,
                                    float* __restrict__ out) {
    __shared__ float sbuf[512];
    const int b = blockIdx.x;
    const int tid = threadIdx.x;
    const float v = fused[(size_t)b * NN + tid];
    sbuf[tid] = v; __syncthreads();
    for (int s = 256; s > 0; s >>= 1) {
        if (tid < s) sbuf[tid] = fmaxf(sbuf[tid], sbuf[tid + s]);
        __syncthreads();
    }
    const float m = sbuf[0]; __syncthreads();
    const float e = expf(v - m);
    sbuf[tid] = e; __syncthreads();
    for (int s = 256; s > 0; s >>= 1) {
        if (tid < s) sbuf[tid] += sbuf[tid + s];
        __syncthreads();
    }
    const float tot = sbuf[0]; __syncthreads();
    const float f = e / tot;
    out[(size_t)b * NN + tid] = f;
    sbuf[tid] = f; __syncthreads();
    for (int s = 256; s > 0; s >>= 1) {
        if (tid < s) sbuf[tid] += sbuf[tid + s];
        __syncthreads();
    }
    if (tid == 0) out[(size_t)BB * NN + b] = sbuf[0] * (1.0f / (float)NN);
}

// ------------------------------- launcher -------------------------------------
extern "C" void kernel_launch(void* const* d_in, const int* in_sizes, int n_in,
                              void* d_out, int out_size, void* d_ws, size_t ws_size,
                              hipStream_t stream) {
    const float* q_emb    = (const float*)d_in[0];
    const float* ctx      = (const float*)d_in[1];
    const float* local_s  = (const float*)d_in[2];
    const float* global_s = (const float*)d_in[3];
    const int*   spans    = (const int*)d_in[4];
    const float* ln_w     = (const float*)d_in[5];
    const float* ln_b     = (const float*)d_in[6];
    const float* w1       = (const float*)d_in[7];
    const float* b1       = (const float*)d_in[8];
    const float* w2       = (const float*)d_in[9];
    const float* b2       = (const float*)d_in[10];
    float* out = (float*)d_out;

    char* ws = (char*)d_ws;
    size_t off = 0;
    auto alloc = [&](size_t bytes) { size_t o = off; off = (off + bytes + 255) & ~(size_t)255; return o; };
    float*          q_vec = (float*)(ws + alloc((size_t)BB * HH * 4));
    float*          qW1   = (float*)(ws + alloc((size_t)BB * HH * 4));
    float*          bs    = (float*)(ws + alloc((size_t)BB * NBLK * HH * 4));
    float*          bp    = (float*)(ws + alloc((size_t)BB * (NBLK + 1) * HH * 4));
    __hip_bfloat16* chunk = (__hip_bfloat16*)(ws + alloc((size_t)BB * NN * HH * 2));
    __hip_bfloat16* w1T   = (__hip_bfloat16*)(ws + alloc((size_t)HH * HH * 2));
    float*          hdn   = (float*)(ws + alloc((size_t)BB * NN * HH * 4));
    float*          fused = (float*)(ws + alloc((size_t)BB * NN * 4));
    (void)ws_size; (void)in_sizes; (void)n_in; (void)out_size;

    mver_qvec_kernel<<<BB, 256, 0, stream>>>(q_emb, ln_w, ln_b, q_vec);
    mver_blocksum_kernel<<<BB * NBLK * 4, 256, 0, stream>>>(ctx, bs);
    mver_blockprefix_kernel<<<BB * 4, 256, 0, stream>>>(bs, bp);
    mver_w1t_kernel<<<(HH * HH) / 256, 256, 0, stream>>>(w1, w1T);
    mver_qw1_kernel<<<BB * 4, 256, 0, stream>>>(q_vec, w1, b1, qW1);
    mver_span_kernel<<<BB * NN, 256, 0, stream>>>(ctx, spans, bp, ln_w, ln_b, chunk);
    mver_gemm_wmma_kernel<<<(BB * NN / 32) * (HH / 64) / 8, 256, 0, stream>>>(chunk, w1T, qW1, hdn);
    mver_gate_kernel<<<BB * NN, 256, 0, stream>>>(hdn, w2, b2, local_s, global_s, fused);
    mver_softmax_kernel<<<BB, NN, 0, stream>>>(fused, out);
}
